// KeyValueMemoryLookup_57758720196678
// MI455X (gfx1250) — matrix operand order repaired and checked by reading, not compiled, not run on hardware
//
#include <hip/hip_runtime.h>

// Problem constants (from reference setup_inputs)
#define BB 16
#define SS 1024
#define MM 2048
#define DD 128
#define DV 128

#define BLOCK_S 64    // S rows per block (4 waves x 16)
#define NWAVES  (BLOCK_S / 16)
#define NTHREADS (NWAVES * 32)
#define MT 32         // memory slots per inner step
#define NSTEPS (MM / MT)

typedef __attribute__((ext_vector_type(16))) _Float16 v16h;
typedef __attribute__((ext_vector_type(8)))  _Float16 v8h;
typedef __attribute__((ext_vector_type(8)))  float    v8f;

// LDS row paddings (in halves): rows stay 16B-aligned (pad%8==0) while the
// row stride mod 256B spreads b128 accesses across bank groups.
#define KPAD 8   // sK  row = 128+8 halves = 272B
#define VPAD 8   // sVt row = 32+8  halves = 80B
#define PPAD 8   // sP  row = 32+8  halves = 80B
#define KSTR (DD + KPAD)
#define VSTR (MT + VPAD)
#define PSTR (MT + PPAD)

__device__ __forceinline__ v16h ld_lds_32B(const _Float16* p) {
  v8h lo = *(const v8h*)p;
  v8h hi = *(const v8h*)(p + 16);
  v16h r;
#pragma unroll
  for (int i = 0; i < 8; ++i) { r[i] = lo[i]; r[i + 8] = hi[i]; }
  return r;
}

// gfx1250 async memory->LDS copy (ASYNCcnt-tracked), GVS addressing:
// mem = saddr(SGPR64) + voff(bytes) ; lds dest = lds_off (low 32 bits of
// generic pointer into __shared__ == wave LDS offset per flat-aperture rule).
__device__ __forceinline__ void async_ld_b128(unsigned lds_off, unsigned voff,
                                              const void* sbase) {
  asm volatile("global_load_async_to_lds_b128 %0, %1, %2"
               :: "v"(lds_off), "v"(voff), "s"(sbase)
               : "memory");
}
__device__ __forceinline__ void wait_async0() {
  asm volatile("s_wait_asynccnt 0x0" ::: "memory");
}
__device__ __forceinline__ unsigned lds_off_of(const void* p) {
  return (unsigned)(uintptr_t)p;
}

// ---------------------------------------------------------------------------
// Prep kernel: L2-normalize K rows -> f16, and transpose V -> f16 [B][Dv][M].
// ---------------------------------------------------------------------------
__global__ __launch_bounds__(256, 1)
void kv_prep_kernel(const float* __restrict__ k_in,
                    const float* __restrict__ v_in,
                    unsigned short* __restrict__ kh_out,   // [B*M][D] f16
                    unsigned short* __restrict__ vt_out) { // [B][Dv][M] f16
  const int tid = threadIdx.x;
  const int r   = blockIdx.x * 32 + (tid >> 3);   // global row in [0, B*M)
  const int seg = (tid & 7) * 16;

  _Float16* kh = (_Float16*)kh_out;
  _Float16* vt = (_Float16*)vt_out;

  {
    const float* kp = k_in + (size_t)r * DD + seg;
    float kb[16];
    float ks = 0.f;
#pragma unroll
    for (int q4 = 0; q4 < 4; ++q4) {
      float4 f = *(const float4*)(kp + q4 * 4);
      kb[q4*4+0] = f.x; kb[q4*4+1] = f.y; kb[q4*4+2] = f.z; kb[q4*4+3] = f.w;
      ks += f.x*f.x + f.y*f.y + f.z*f.z + f.w*f.w;
    }
    ks += __shfl_xor(ks, 1, 32);
    ks += __shfl_xor(ks, 2, 32);
    ks += __shfl_xor(ks, 4, 32);
    const float kinv = 1.f / fmaxf(sqrtf(ks), 1e-12f);
    _Float16* kd = kh + (size_t)r * DD + seg;
#pragma unroll
    for (int i = 0; i < 16; ++i) kd[i] = (_Float16)(kb[i] * kinv);
  }
  {
    const int b = r >> 11;
    const int m = r & (MM - 1);
    const float* vp = v_in + (size_t)r * DD + seg;
#pragma unroll
    for (int q4 = 0; q4 < 4; ++q4) {
      float4 f = *(const float4*)(vp + q4 * 4);
      const int d0 = seg + q4 * 4;
      vt[((size_t)b * DV + d0 + 0) * MM + m] = (_Float16)f.x;
      vt[((size_t)b * DV + d0 + 1) * MM + m] = (_Float16)f.y;
      vt[((size_t)b * DV + d0 + 2) * MM + m] = (_Float16)f.z;
      vt[((size_t)b * DV + d0 + 3) * MM + m] = (_Float16)f.w;
    }
  }
}

// ---------------------------------------------------------------------------
// Shared Q-fragment setup and per-tile compute body.
// ---------------------------------------------------------------------------
__device__ __forceinline__ void build_q_frags(const float* __restrict__ q_in,
                                              int b, int s_row0, int lo16, int hi,
                                              v16h qa[4]) {
  const float* qrow = q_in + ((size_t)b * SS + s_row0 + lo16) * DD;
  const int hi8 = hi * 8;
  float qbuf[64];
  float sumsq = 0.f;
#pragma unroll
  for (int c = 0; c < 4; ++c) {
#pragma unroll
    for (int r = 0; r < 2; ++r) {
      const int d0 = 32 * c + 16 * r + hi8;
      float4 f0 = *(const float4*)(qrow + d0);
      float4 f1 = *(const float4*)(qrow + d0 + 4);
      float* dst = &qbuf[c * 16 + r * 8];
      dst[0] = f0.x; dst[1] = f0.y; dst[2] = f0.z; dst[3] = f0.w;
      dst[4] = f1.x; dst[5] = f1.y; dst[6] = f1.z; dst[7] = f1.w;
      sumsq += f0.x*f0.x + f0.y*f0.y + f0.z*f0.z + f0.w*f0.w
             + f1.x*f1.x + f1.y*f1.y + f1.z*f1.z + f1.w*f1.w;
    }
  }
  sumsq += __shfl_xor(sumsq, 16, 32);
  const float qinv = 1.f / fmaxf(sqrtf(sumsq), 1e-12f);
#pragma unroll
  for (int c = 0; c < 4; ++c)
#pragma unroll
    for (int h = 0; h < 16; ++h)
      qa[c][h] = (_Float16)(qbuf[c * 16 + h] * qinv);
}

__device__ __forceinline__ void attn_step(const _Float16* __restrict__ sKp,
                                          const _Float16* __restrict__ sVp,
                                          _Float16* __restrict__ sP_w,
                                          const v16h qa[4], v8f oacc[8],
                                          float psum[8], int lo16, int hi) {
  v8f sc0 = v8f{}, sc1 = v8f{};
#pragma unroll
  for (int c = 0; c < 4; ++c) {
    v16h kb0 = ld_lds_32B(sKp + lo16 * KSTR + 32 * c + 8 * hi);
    sc0 = __builtin_amdgcn_wmma_f32_16x16x32_f16(false, qa[c], false, kb0,
                                                 (short)0, sc0, false, false);
  }
#pragma unroll
  for (int c = 0; c < 4; ++c) {
    v16h kb1 = ld_lds_32B(sKp + (16 + lo16) * KSTR + 32 * c + 8 * hi);
    sc1 = __builtin_amdgcn_wmma_f32_16x16x32_f16(false, qa[c], false, kb1,
                                                 (short)0, sc1, false, false);
  }

  // exp without max shift: cosine scores are bounded in [-1, 1].
#pragma unroll
  for (int v = 0; v < 8; ++v) {
    const float p0 = __expf(sc0[v]);
    const float p1 = __expf(sc1[v]);
    const _Float16 h0 = (_Float16)p0;
    const _Float16 h1 = (_Float16)p1;
    psum[v] += (float)h0 + (float)h1;
    const int m = v + 8 * hi;
    sP_w[m * PSTR + lo16]      = h0;
    sP_w[m * PSTR + 16 + lo16] = h1;
  }

  v8h plo = *(const v8h*)&sP_w[lo16 * PSTR + 8 * hi];
  v8h phi = *(const v8h*)&sP_w[lo16 * PSTR + 8 * hi + 16];
  v16h pa;
#pragma unroll
  for (int i = 0; i < 8; ++i) { pa[i] = plo[i]; pa[i + 8] = phi[i]; }
#pragma unroll
  for (int nt = 0; nt < 8; ++nt) {
    v16h vb = ld_lds_32B(sVp + (nt * 16 + lo16) * VSTR + 8 * hi);
    oacc[nt] = __builtin_amdgcn_wmma_f32_16x16x32_f16(false, pa, false, vb,
                                                      (short)0, oacc[nt],
                                                      false, false);
  }
}

__device__ __forceinline__ void finalize(float* __restrict__ out, int b,
                                         int s_row0, int lo16, int hi,
                                         v8f oacc[8], float psum[8]) {
#pragma unroll
  for (int v = 0; v < 8; ++v) {
    float rs = psum[v];
    rs += __shfl_xor(rs, 1, 32);
    rs += __shfl_xor(rs, 2, 32);
    rs += __shfl_xor(rs, 4, 32);
    rs += __shfl_xor(rs, 8, 32);
    const float rl = 1.0f / rs;
    const int srow = s_row0 + v + 8 * hi;
    float* op = out + ((size_t)b * SS + srow) * DV + lo16;
#pragma unroll
    for (int nt = 0; nt < 8; ++nt)
      op[nt * 16] = oacc[nt][v] * rl;
  }
}

// ---------------------------------------------------------------------------
// Fast path: pre-normalized f16 K / pre-transposed f16 V in workspace,
// async global->LDS tile copies, double-buffered, one barrier per step.
// ---------------------------------------------------------------------------
__global__ __launch_bounds__(NTHREADS, 1)
void kv_attn_async_kernel(const float* __restrict__ q_in,
                          const unsigned short* __restrict__ kh_in,
                          const unsigned short* __restrict__ vt_in,
                          float* __restrict__ out) {
  __shared__ __align__(16) _Float16 sK2[2][MT][KSTR];
  __shared__ __align__(16) _Float16 sV2[2][DV][VSTR];
  __shared__ __align__(16) _Float16 sP[NWAVES][16][PSTR];

  const int tid  = threadIdx.x;
  const int wave = tid >> 5;
  const int lane = tid & 31;
  const int lo16 = lane & 15;
  const int hi   = lane >> 4;

  const int sblocks = SS / BLOCK_S;
  const int b  = blockIdx.x / sblocks;
  const int sb = blockIdx.x % sblocks;
  const int s_row0 = sb * BLOCK_S + wave * 16;

  v16h qa[4];
  build_q_frags(q_in, b, s_row0, lo16, hi, qa);

  v8f oacc[8];
#pragma unroll
  for (int nt = 0; nt < 8; ++nt) oacc[nt] = v8f{};
  float psum[8];
#pragma unroll
  for (int v = 0; v < 8; ++v) psum[v] = 0.f;
  _Float16* sP_w = &sP[wave][0][0];

  // Per-thread tile-copy assignment:
  //   K: 4 threads/row, 32 halves (64B) each -> 4 async b128
  //   V: 1 dv row/thread, 32 halves (64B)    -> 4 async b128
  const int krow = tid >> 2;
  const int kseg = (tid & 3) * 32;
  const _Float16* k_base = (const _Float16*)kh_in + (size_t)b * MM * DD;
  const _Float16* v_base = (const _Float16*)vt_in + (size_t)b * DV * MM;
  const unsigned k_voff0 = (unsigned)((krow * DD + kseg) * 2);      // bytes
  const unsigned v_voff0 = (unsigned)(tid * MM * 2);                // bytes
  unsigned kL[2], vL[2];
#pragma unroll
  for (int u = 0; u < 2; ++u) {
    kL[u] = lds_off_of(&sK2[u][krow][kseg]);
    vL[u] = lds_off_of(&sV2[u][tid][0]);
  }

#define ISSUE_TILES(it, buf)                                                  \
  do {                                                                        \
    const unsigned kv = k_voff0 + (unsigned)(it) * (MT * DD * 2);             \
    const unsigned vv = v_voff0 + (unsigned)(it) * (MT * 2);                  \
    _Pragma("unroll")                                                         \
    for (int q4 = 0; q4 < 4; ++q4) {                                          \
      async_ld_b128(kL[buf] + q4 * 16, kv + q4 * 16, k_base);                 \
      async_ld_b128(vL[buf] + q4 * 16, vv + q4 * 16, v_base);                 \
    }                                                                         \
  } while (0)

  ISSUE_TILES(0, 0);

  for (int it = 0; it < NSTEPS; ++it) {
    const int cur = it & 1;
    wait_async0();       // this wave's tile `it` landed in LDS
    __syncthreads();     // everyone's tile `it` visible; buf cur^1 free
    if (it + 1 < NSTEPS) ISSUE_TILES(it + 1, cur ^ 1);
    attn_step(&sK2[cur][0][0], &sV2[cur][0][0], sP_w, qa, oacc, psum, lo16, hi);
  }
#undef ISSUE_TILES

  finalize(out, b, s_row0, lo16, hi, oacc, psum);
}

// ---------------------------------------------------------------------------
// Fallback: fused normalize/convert from f32 inputs (no workspace needed).
// ---------------------------------------------------------------------------
__global__ __launch_bounds__(NTHREADS, 1)
void kv_attn_fused_kernel(const float* __restrict__ q_in,
                          const float* __restrict__ k_in,
                          const float* __restrict__ v_in,
                          float* __restrict__ out) {
  __shared__ __align__(16) _Float16 sK[MT][KSTR];
  __shared__ __align__(16) _Float16 sVt[DV][VSTR];
  __shared__ __align__(16) _Float16 sP[NWAVES][16][PSTR];

  const int tid  = threadIdx.x;
  const int wave = tid >> 5;
  const int lane = tid & 31;
  const int lo16 = lane & 15;
  const int hi   = lane >> 4;

  const int sblocks = SS / BLOCK_S;
  const int b  = blockIdx.x / sblocks;
  const int sb = blockIdx.x % sblocks;
  const int s_row0 = sb * BLOCK_S + wave * 16;

  v16h qa[4];
  build_q_frags(q_in, b, s_row0, lo16, hi, qa);

  v8f oacc[8];
#pragma unroll
  for (int nt = 0; nt < 8; ++nt) oacc[nt] = v8f{};
  float psum[8];
#pragma unroll
  for (int v = 0; v < 8; ++v) psum[v] = 0.f;
  _Float16* sP_w = &sP[wave][0][0];

  for (int mt0 = 0; mt0 < MM; mt0 += MT) {
    __syncthreads();
    {
      const int krow = tid >> 2;
      const int kseg = (tid & 3) * 32;
      const float* kp = k_in + ((size_t)b * MM + mt0 + krow) * DD + kseg;
      float kb[32];
      float ks = 0.f;
#pragma unroll
      for (int q4 = 0; q4 < 8; ++q4) {
        float4 f = *(const float4*)(kp + q4 * 4);
        kb[q4*4+0] = f.x; kb[q4*4+1] = f.y; kb[q4*4+2] = f.z; kb[q4*4+3] = f.w;
        ks += f.x*f.x + f.y*f.y + f.z*f.z + f.w*f.w;
      }
      ks += __shfl_xor(ks, 1, 32);
      ks += __shfl_xor(ks, 2, 32);
      const float kinv = 1.f / fmaxf(sqrtf(ks), 1e-12f);
#pragma unroll
      for (int i = 0; i < 32; ++i)
        sK[krow][kseg + i] = (_Float16)(kb[i] * kinv);
    }
    {
      const int vrow = tid >> 2;
      const int vseg = (tid & 3) * 32;
      const float* vp = v_in + ((size_t)b * MM + mt0 + vrow) * DD + vseg;
#pragma unroll
      for (int q4 = 0; q4 < 8; ++q4) {
        float4 f = *(const float4*)(vp + q4 * 4);
        sVt[vseg + q4*4 + 0][vrow] = (_Float16)f.x;
        sVt[vseg + q4*4 + 1][vrow] = (_Float16)f.y;
        sVt[vseg + q4*4 + 2][vrow] = (_Float16)f.z;
        sVt[vseg + q4*4 + 3][vrow] = (_Float16)f.w;
      }
    }
    __syncthreads();
    attn_step(&sK[0][0], &sVt[0][0], sP_w, qa, oacc, psum, lo16, hi);
  }

  finalize(out, b, s_row0, lo16, hi, oacc, psum);
}

extern "C" void kernel_launch(void* const* d_in, const int* in_sizes, int n_in,
                              void* d_out, int out_size, void* d_ws, size_t ws_size,
                              hipStream_t stream) {
  (void)in_sizes; (void)n_in; (void)out_size;
  const float* q = (const float*)d_in[0];
  const float* k = (const float*)d_in[1];
  const float* v = (const float*)d_in[2];
  float* out = (float*)d_out;

  const size_t kh_bytes = (size_t)BB * MM * DD * sizeof(unsigned short);
  const size_t vt_bytes = (size_t)BB * DV * MM * sizeof(unsigned short);

  dim3 grid(BB * (SS / BLOCK_S));   // 256 blocks
  dim3 block(NTHREADS);             // 4 wave32 waves

  if (ws_size >= kh_bytes + vt_bytes) {
    unsigned short* kh = (unsigned short*)d_ws;
    unsigned short* vt = (unsigned short*)((char*)d_ws + kh_bytes);
    kv_prep_kernel<<<dim3((BB * MM) / 32), dim3(256), 0, stream>>>(k, v, kh, vt);
    kv_attn_async_kernel<<<grid, block, 0, stream>>>(q, kh, vt, out);
  } else {
    kv_attn_fused_kernel<<<grid, block, 0, stream>>>(q, k, v, out);
  }
}
